// MultiHeadSelfAttention_18648747999536
// MI455X (gfx1250) — matrix operand order repaired and checked by reading, not compiled, not run on hardware
//
#include <hip/hip_runtime.h>
#include <hip/hip_bf16.h>

// ---------------- problem constants ----------------
constexpr int Bb     = 16;
constexpr int Nn     = 1024;
constexpr int Cc     = 768;
constexpr int Hh     = 12;
constexpr int Dd     = 64;
constexpr int QKVC   = 3 * Cc;              // 2304
constexpr float SCALE = 0.125f;             // 64^-0.5
constexpr size_t HEAD_ELEMS = (size_t)Bb * Hh * Nn * Dd;   // per Q/K/V buf

typedef __attribute__((ext_vector_type(16))) __bf16 v16bf;
typedef __attribute__((ext_vector_type(8)))  float  v8f;

union AFrag { v16bf v; uint4 q[2]; };
union U8h   { uint4 q; __bf16 h[8]; };

#define WMMA_BF16(a, b, c) \
  __builtin_amdgcn_wmma_f32_16x16x32_bf16(false, (a), false, (b), (short)0, (c), false, false)

static __device__ __forceinline__ v8f vzero8() {
  v8f z = {0.f, 0.f, 0.f, 0.f, 0.f, 0.f, 0.f, 0.f};
  return z;
}

// pack 4 floats -> 4 bf16 -> one 8-byte LDS store
static __device__ __forceinline__ void store4bf(__bf16* dst, float a, float b,
                                                float c, float d) {
  union { __bf16 h[4]; uint2 u; } t;
  t.h[0] = (__bf16)a; t.h[1] = (__bf16)b; t.h[2] = (__bf16)c; t.h[3] = (__bf16)d;
  *(uint2*)dst = t.u;
}

// ---------------- TDM (Tensor Data Mover) fast path ----------------
#if __has_builtin(__builtin_amdgcn_tensor_load_to_lds)
#define HAVE_TDM 1
typedef __attribute__((ext_vector_type(4))) unsigned int u32x4;
typedef __attribute__((ext_vector_type(8))) int          i32x8;
typedef __attribute__((ext_vector_type(4))) int          i32x4;

// DMA a 64x64 bf16 tile (row stride 64 elems in global) into LDS with
// 4-DWORD padding every 32 DWORDs (-> LDS row stride 72 elems).
static __device__ __forceinline__ void tdm_load_tile_64x64(unsigned ldsByteAddr,
                                                           const void* gsrc) {
  unsigned long long ga = (unsigned long long)gsrc;
  u32x4 g0;
  g0[0] = 1u;                                             // count=1, user mode
  g0[1] = ldsByteAddr;                                    // lds_addr
  g0[2] = (unsigned)(ga & 0xffffffffu);                   // global_addr[31:0]
  g0[3] = (unsigned)((ga >> 32) & 0x01ffffffu) | (2u << 30); // ga[56:32] | type=2
  i32x8 g1;
  g1[0] = (1 << 16)      // data_size = 2 bytes
        | (1 << 20)      // pad_enable
        | (4 << 22)      // pad_interval: every 32 DWORDs
        | (3 << 25);     // pad_amount: 4 DWORDs
  g1[1] = (64 & 0xffff) << 16;   // tensor_dim0 lo16 (64)
  g1[2] = (64 << 16);            // tensor_dim0 hi | tensor_dim1 lo (64)
  g1[3] = (64 << 16);            // tensor_dim1 hi | tile_dim0 = 64
  g1[4] = 64;                    // tile_dim1 = 64, tile_dim2 = 0
  g1[5] = 64;                    // tensor_dim0_stride lo32 = 64
  g1[6] = 0;
  g1[7] = 0;
  i32x4 z4 = {0, 0, 0, 0};
#if defined(__clang_major__) && (__clang_major__ >= 23)
  i32x8 z8 = {0, 0, 0, 0, 0, 0, 0, 0};
  __builtin_amdgcn_tensor_load_to_lds(g0, g1, z4, z4, z8, 0);
#else
  __builtin_amdgcn_tensor_load_to_lds(g0, g1, z4, z4, 0);
#endif
}
#endif

// =====================================================================
// Kernel 1: QKV projection.  X[16384,768] (f32) @ Wqkv[768,2304] (f32)
// -> Q/K/V bf16 buffers in [B,H,N,D] layout.
// Block tile 128(M) x 128(N), 8 waves of 2x4 16x16 WMMA tiles, K-step 32.
// =====================================================================
__global__ __launch_bounds__(256) void qkv_gemm_kernel(
    const float* __restrict__ x, const float* __restrict__ w,
    unsigned short* __restrict__ qraw, unsigned short* __restrict__ kraw,
    unsigned short* __restrict__ vraw)
{
  __shared__ __align__(16) __bf16 Xs[128 * 40];   // [row][k]  stride 40
  __shared__ __align__(16) __bf16 Ws[128 * 40];   // [col][k]  stride 40 (transposed)

  __bf16* qb = (__bf16*)qraw;
  __bf16* kb = (__bf16*)kraw;
  __bf16* vb = (__bf16*)vraw;

  const int tid  = threadIdx.x;
  const int lane = tid & 31;
  const int wave = tid >> 5;
  const int hi   = lane >> 4;
  const int lo   = lane & 15;
  const int wm   = wave >> 1;   // 0..3 : 32-row group
  const int wn   = wave & 1;    // 0..1 : 64-col group
  const int m0   = blockIdx.y * 128;
  const int n0   = blockIdx.x * 128;

  const int wc4 = (tid & 31) * 4;   // W staging: col offset 0..124
  const int wk  = (tid >> 5) * 4;   // W staging: k row 0..28

  v8f acc[2][4];
#pragma unroll
  for (int r = 0; r < 2; ++r)
#pragma unroll
    for (int t = 0; t < 4; ++t) acc[r][t] = vzero8();

  for (int k0 = 0; k0 < Cc; k0 += 32) {
    __syncthreads();
    // ---- load phase (batched global loads) ----
    float4 xv0, xv1, xv2, xv3;
    {
      int i0 = tid, i1 = tid + 256, i2 = tid + 512, i3 = tid + 768;
      xv0 = *(const float4*)(x + (size_t)(m0 + (i0 >> 3)) * Cc + k0 + (i0 & 7) * 4);
      xv1 = *(const float4*)(x + (size_t)(m0 + (i1 >> 3)) * Cc + k0 + (i1 & 7) * 4);
      xv2 = *(const float4*)(x + (size_t)(m0 + (i2 >> 3)) * Cc + k0 + (i2 & 7) * 4);
      xv3 = *(const float4*)(x + (size_t)(m0 + (i3 >> 3)) * Cc + k0 + (i3 & 7) * 4);
    }
    float4 wv0 = *(const float4*)(w + (size_t)(k0 + wk + 0) * QKVC + n0 + wc4);
    float4 wv1 = *(const float4*)(w + (size_t)(k0 + wk + 1) * QKVC + n0 + wc4);
    float4 wv2 = *(const float4*)(w + (size_t)(k0 + wk + 2) * QKVC + n0 + wc4);
    float4 wv3 = *(const float4*)(w + (size_t)(k0 + wk + 3) * QKVC + n0 + wc4);

    // ---- store phase: X natural layout (k-contiguous b64 stores) ----
    {
      int i0 = tid, i1 = tid + 256, i2 = tid + 512, i3 = tid + 768;
      store4bf(&Xs[(i0 >> 3) * 40 + (i0 & 7) * 4], xv0.x, xv0.y, xv0.z, xv0.w);
      store4bf(&Xs[(i1 >> 3) * 40 + (i1 & 7) * 4], xv1.x, xv1.y, xv1.z, xv1.w);
      store4bf(&Xs[(i2 >> 3) * 40 + (i2 & 7) * 4], xv2.x, xv2.y, xv2.z, xv2.w);
      store4bf(&Xs[(i3 >> 3) * 40 + (i3 & 7) * 4], xv3.x, xv3.y, xv3.z, xv3.w);
    }
    // ---- store phase: W transposed, 4 k-rows packed per b64 store ----
    store4bf(&Ws[(wc4 + 0) * 40 + wk], wv0.x, wv1.x, wv2.x, wv3.x);
    store4bf(&Ws[(wc4 + 1) * 40 + wk], wv0.y, wv1.y, wv2.y, wv3.y);
    store4bf(&Ws[(wc4 + 2) * 40 + wk], wv0.z, wv1.z, wv2.z, wv3.z);
    store4bf(&Ws[(wc4 + 3) * 40 + wk], wv0.w, wv1.w, wv2.w, wv3.w);
    __syncthreads();

    AFrag a[2], b[4];
#pragma unroll
    for (int r = 0; r < 2; ++r) {
      const __bf16* base = &Xs[(wm * 32 + r * 16 + lo) * 40];
      a[r].q[0] = *(const uint4*)(base + hi * 8);
      a[r].q[1] = *(const uint4*)(base + 16 + hi * 8);
    }
#pragma unroll
    for (int t = 0; t < 4; ++t) {
      const __bf16* base = &Ws[(wn * 64 + t * 16 + lo) * 40];
      b[t].q[0] = *(const uint4*)(base + hi * 8);
      b[t].q[1] = *(const uint4*)(base + 16 + hi * 8);
    }
#pragma unroll
    for (int r = 0; r < 2; ++r)
#pragma unroll
      for (int t = 0; t < 4; ++t)
        acc[r][t] = WMMA_BF16(a[r].v, b[t].v, acc[r][t]);
  }

  // ---- scatter store into Q/K/V [B,H,N,D] bf16 ----
#pragma unroll
  for (int r = 0; r < 2; ++r) {
#pragma unroll
    for (int t = 0; t < 4; ++t) {
      int col  = n0 + wn * 64 + t * 16 + lo;      // 0..2303
      int sect = col / Cc;                        // 0=q 1=k 2=v
      int rem  = col - sect * Cc;
      int h    = rem >> 6;
      int d    = rem & 63;
      __bf16* dst = (sect == 0) ? qb : ((sect == 1) ? kb : vb);
#pragma unroll
      for (int v = 0; v < 8; ++v) {
        int m  = m0 + wm * 32 + r * 16 + hi * 8 + v;
        int bi = m >> 10;
        int n  = m & 1023;
        dst[(((size_t)(bi * Hh + h)) * Nn + n) * Dd + d] = (__bf16)acc[r][t][v];
      }
    }
  }
}

// =====================================================================
// Kernel 2: flash attention per (b,h).  Block = 128 query rows, 8 waves
// of 16 rows each.  Streams keys/values in 64-chunks through LDS with
// online softmax.  K chunk staged via TDM when available.
// =====================================================================
__global__ __launch_bounds__(256) void attn_kernel(
    const unsigned short* __restrict__ qraw, const unsigned short* __restrict__ kraw,
    const unsigned short* __restrict__ vraw, unsigned short* __restrict__ araw)
{
  __shared__ __align__(16) __bf16 Ks[64 * 72];        // [key][d]   stride 72
  __shared__ __align__(16) __bf16 Vs[64 * 72];        // [d][key]   stride 72 (transposed)
  __shared__ __align__(16) __bf16 Ps[8][16 * 72];     // per-wave P tile [row][key]

  const __bf16* qb = (const __bf16*)qraw;
  const __bf16* kb = (const __bf16*)kraw;
  const __bf16* vb = (const __bf16*)vraw;
  __bf16* ab = (__bf16*)araw;

  const int tid  = threadIdx.x;
  const int lane = tid & 31;
  const int wave = tid >> 5;
  const int hi   = lane >> 4;
  const int lo   = lane & 15;
  const int bh   = blockIdx.y;                 // 0..191
  const int qt   = blockIdx.x;                 // 0..7
  const size_t headBase = (size_t)bh * Nn * Dd;

  // ---- resident Q fragments ----
  AFrag qa[2];
  {
    const __bf16* qp = qb + headBase + (size_t)(qt * 128 + wave * 16 + lo) * Dd;
    qa[0].q[0] = *(const uint4*)(qp + hi * 8);
    qa[0].q[1] = *(const uint4*)(qp + 16 + hi * 8);
    qa[1].q[0] = *(const uint4*)(qp + 32 + hi * 8);
    qa[1].q[1] = *(const uint4*)(qp + 48 + hi * 8);
  }

  v8f o[4];
#pragma unroll
  for (int dt = 0; dt < 4; ++dt) o[dt] = vzero8();
  float mrun[8], lrun[8];
#pragma unroll
  for (int v = 0; v < 8; ++v) { mrun[v] = -1e30f; lrun[v] = 0.f; }

  const int vg = tid >> 3;           // key pair id 0..31
  const int vc = (tid & 7) * 8;      // d offset
#ifdef HAVE_TDM
  const unsigned ldsK = (unsigned)(uintptr_t)&Ks[0];
#endif

  for (int kc = 0; kc < Nn; kc += 64) {
    __syncthreads();
#ifdef HAVE_TDM
    // ---- K chunk via Tensor Data Mover (64x64 bf16, padded to stride 72) ----
    if (wave == 0) {
      tdm_load_tile_64x64(ldsK, kb + headBase + (size_t)kc * Dd);
      __builtin_amdgcn_s_wait_tensorcnt(0);
    }
#else
    // ---- K chunk: batched vector copy ----
    {
      int i0 = tid, i1 = tid + 256;
      uint4 k0v = *(const uint4*)(kb + headBase + (size_t)(kc + (i0 >> 3)) * Dd + (i0 & 7) * 8);
      uint4 k1v = *(const uint4*)(kb + headBase + (size_t)(kc + (i1 >> 3)) * Dd + (i1 & 7) * 8);
      *(uint4*)(&Ks[(i0 >> 3) * 72 + (i0 & 7) * 8]) = k0v;
      *(uint4*)(&Ks[(i1 >> 3) * 72 + (i1 & 7) * 8]) = k1v;
    }
#endif
    // ---- V chunk transposed -> Vs[d][key]; 2 keys packed per b32 store ----
    {
      U8h r0, r1;
      r0.q = *(const uint4*)(vb + headBase + (size_t)(kc + 2 * vg + 0) * Dd + vc);
      r1.q = *(const uint4*)(vb + headBase + (size_t)(kc + 2 * vg + 1) * Dd + vc);
#pragma unroll
      for (int j = 0; j < 8; ++j) {
        union { __bf16 h[2]; unsigned u; } p;
        p.h[0] = r0.h[j]; p.h[1] = r1.h[j];
        *(unsigned*)(&Vs[(vc + j) * 72 + 2 * vg]) = p.u;
      }
    }
    __syncthreads();

    // ---- S = Q K^T  (4 tiles of 16 keys, contraction D=64 in 2 steps) ----
    v8f s[4];
#pragma unroll
    for (int nt = 0; nt < 4; ++nt) {
      AFrag kb0, kb1;
      const __bf16* base = &Ks[(nt * 16 + lo) * 72];
      kb0.q[0] = *(const uint4*)(base + hi * 8);
      kb0.q[1] = *(const uint4*)(base + 16 + hi * 8);
      kb1.q[0] = *(const uint4*)(base + 32 + hi * 8);
      kb1.q[1] = *(const uint4*)(base + 48 + hi * 8);
      v8f t = vzero8();
      t = WMMA_BF16(qa[0].v, kb0.v, t);
      t = WMMA_BF16(qa[1].v, kb1.v, t);
      s[nt] = t * SCALE;
    }

    // ---- online softmax update (rows live in 16-lane halves) ----
#pragma unroll
    for (int v = 0; v < 8; ++v) {
      float mx = fmaxf(fmaxf(s[0][v], s[1][v]), fmaxf(s[2][v], s[3][v]));
#pragma unroll
      for (int off = 1; off < 16; off <<= 1) mx = fmaxf(mx, __shfl_xor(mx, off, 32));
      float mnew  = fmaxf(mrun[v], mx);
      float alpha = __expf(mrun[v] - mnew);
      mrun[v] = mnew;
      float rs = 0.f;
#pragma unroll
      for (int nt = 0; nt < 4; ++nt) {
        float p = __expf(s[nt][v] - mnew);
        rs += p;
        Ps[wave][(hi * 8 + v) * 72 + nt * 16 + lo] = (__bf16)p;
      }
#pragma unroll
      for (int off = 1; off < 16; off <<= 1) rs += __shfl_xor(rs, off, 32);
      lrun[v] = lrun[v] * alpha + rs;
#pragma unroll
      for (int dt = 0; dt < 4; ++dt) o[dt][v] *= alpha;
    }

    // ---- O += P @ V  (per-wave P from LDS, contraction 64 keys) ----
    AFrag pa[2];
    {
      const __bf16* pbase = &Ps[wave][lo * 72];
      pa[0].q[0] = *(const uint4*)(pbase + hi * 8);
      pa[0].q[1] = *(const uint4*)(pbase + 16 + hi * 8);
      pa[1].q[0] = *(const uint4*)(pbase + 32 + hi * 8);
      pa[1].q[1] = *(const uint4*)(pbase + 48 + hi * 8);
    }
#pragma unroll
    for (int dt = 0; dt < 4; ++dt) {
      AFrag vb0, vb1;
      const __bf16* vbase = &Vs[(dt * 16 + lo) * 72];
      vb0.q[0] = *(const uint4*)(vbase + hi * 8);
      vb0.q[1] = *(const uint4*)(vbase + 16 + hi * 8);
      vb1.q[0] = *(const uint4*)(vbase + 32 + hi * 8);
      vb1.q[1] = *(const uint4*)(vbase + 48 + hi * 8);
      o[dt] = WMMA_BF16(pa[0].v, vb0.v, o[dt]);
      o[dt] = WMMA_BF16(pa[1].v, vb1.v, o[dt]);
    }
  }

  // ---- epilogue: O /= l, store to Abuf[b][n][h*64+d] (bf16) ----
  const int h = bh % Hh;
  const int b = bh / Hh;
#pragma unroll
  for (int dt = 0; dt < 4; ++dt) {
#pragma unroll
    for (int v = 0; v < 8; ++v) {
      int m = qt * 128 + wave * 16 + hi * 8 + v;
      float val = o[dt][v] / lrun[v];
      ab[((size_t)b * Nn + m) * Cc + h * Dd + dt * 16 + lo] = (__bf16)val;
    }
  }
}

// =====================================================================
// Kernel 3: output projection.  Abuf[16384,768] (bf16) @ Wproj[768,768]
// (f32->bf16) + bias -> out f32.  Block tile 128x128.
// =====================================================================
__global__ __launch_bounds__(256) void proj_gemm_kernel(
    const unsigned short* __restrict__ araw, const float* __restrict__ w,
    const float* __restrict__ bias, float* __restrict__ out)
{
  __shared__ __align__(16) __bf16 As[128 * 40];
  __shared__ __align__(16) __bf16 Ws[128 * 40];

  const __bf16* a = (const __bf16*)araw;

  const int tid  = threadIdx.x;
  const int lane = tid & 31;
  const int wave = tid >> 5;
  const int hi   = lane >> 4;
  const int lo   = lane & 15;
  const int wm   = wave >> 1;
  const int wn   = wave & 1;
  const int m0   = blockIdx.y * 128;
  const int n0   = blockIdx.x * 128;

  const int wc4 = (tid & 31) * 4;
  const int wk  = (tid >> 5) * 4;

  v8f acc[2][4];
#pragma unroll
  for (int r = 0; r < 2; ++r)
#pragma unroll
    for (int t = 0; t < 4; ++t) acc[r][t] = vzero8();

  for (int k0 = 0; k0 < Cc; k0 += 32) {
    __syncthreads();
    // ---- load phase ----
    uint4 av0, av1;
    {
      int i0 = tid, i1 = tid + 256;
      av0 = *(const uint4*)(a + (size_t)(m0 + (i0 >> 2)) * Cc + k0 + (i0 & 3) * 8);
      av1 = *(const uint4*)(a + (size_t)(m0 + (i1 >> 2)) * Cc + k0 + (i1 & 3) * 8);
    }
    float4 wv0 = *(const float4*)(w + (size_t)(k0 + wk + 0) * Cc + n0 + wc4);
    float4 wv1 = *(const float4*)(w + (size_t)(k0 + wk + 1) * Cc + n0 + wc4);
    float4 wv2 = *(const float4*)(w + (size_t)(k0 + wk + 2) * Cc + n0 + wc4);
    float4 wv3 = *(const float4*)(w + (size_t)(k0 + wk + 3) * Cc + n0 + wc4);

    // ---- store phase ----
    {
      int i0 = tid, i1 = tid + 256;
      *(uint4*)(&As[(i0 >> 2) * 40 + (i0 & 3) * 8]) = av0;
      *(uint4*)(&As[(i1 >> 2) * 40 + (i1 & 3) * 8]) = av1;
    }
    store4bf(&Ws[(wc4 + 0) * 40 + wk], wv0.x, wv1.x, wv2.x, wv3.x);
    store4bf(&Ws[(wc4 + 1) * 40 + wk], wv0.y, wv1.y, wv2.y, wv3.y);
    store4bf(&Ws[(wc4 + 2) * 40 + wk], wv0.z, wv1.z, wv2.z, wv3.z);
    store4bf(&Ws[(wc4 + 3) * 40 + wk], wv0.w, wv1.w, wv2.w, wv3.w);
    __syncthreads();

    AFrag af[2], bf[4];
#pragma unroll
    for (int r = 0; r < 2; ++r) {
      const __bf16* base = &As[(wm * 32 + r * 16 + lo) * 40];
      af[r].q[0] = *(const uint4*)(base + hi * 8);
      af[r].q[1] = *(const uint4*)(base + 16 + hi * 8);
    }
#pragma unroll
    for (int t = 0; t < 4; ++t) {
      const __bf16* base = &Ws[(wn * 64 + t * 16 + lo) * 40];
      bf[t].q[0] = *(const uint4*)(base + hi * 8);
      bf[t].q[1] = *(const uint4*)(base + 16 + hi * 8);
    }
#pragma unroll
    for (int r = 0; r < 2; ++r)
#pragma unroll
      for (int t = 0; t < 4; ++t)
        acc[r][t] = WMMA_BF16(af[r].v, bf[t].v, acc[r][t]);
  }

  // ---- store f32 + bias ----
#pragma unroll
  for (int r = 0; r < 2; ++r) {
#pragma unroll
    for (int t = 0; t < 4; ++t) {
      int col = n0 + wn * 64 + t * 16 + lo;
      float bv = bias[col];
#pragma unroll
      for (int v = 0; v < 8; ++v) {
        int m = m0 + wm * 32 + r * 16 + hi * 8 + v;
        out[(size_t)m * Cc + col] = acc[r][t][v] + bv;
      }
    }
  }
}

// =====================================================================
extern "C" void kernel_launch(void* const* d_in, const int* in_sizes, int n_in,
                              void* d_out, int out_size, void* d_ws, size_t ws_size,
                              hipStream_t stream) {
  (void)in_sizes; (void)n_in; (void)out_size; (void)ws_size;
  const float* x      = (const float*)d_in[0];
  const float* w_qkv  = (const float*)d_in[1];
  const float* w_proj = (const float*)d_in[2];
  const float* b_proj = (const float*)d_in[3];
  float* out = (float*)d_out;

  unsigned short* qb = (unsigned short*)d_ws;
  unsigned short* kb = qb + HEAD_ELEMS;
  unsigned short* vb = kb + HEAD_ELEMS;
  unsigned short* ab = vb + HEAD_ELEMS;   // [B,N,768] bf16

  // QKV projection: M=16384, N=2304, 128x128 tiles
  qkv_gemm_kernel<<<dim3(QKVC / 128, (Bb * Nn) / 128), 256, 0, stream>>>(
      x, w_qkv, qb, kb, vb);

  // Flash attention: 8 query tiles x 192 (b,h) heads
  attn_kernel<<<dim3(Nn / 128, Bb * Hh), 256, 0, stream>>>(qb, kb, vb, ab);

  // Output projection: M=16384, N=768, 128x128 tiles
  proj_gemm_kernel<<<dim3(Cc / 128, (Bb * Nn) / 128), 256, 0, stream>>>(
      ab, w_proj, b_proj, out);
}
